// MultiHeadAttention_51616916963549
// MI455X (gfx1250) — matrix operand order repaired and checked by reading, not compiled, not run on hardware
//
#include <hip/hip_runtime.h>

typedef __attribute__((ext_vector_type(16))) _Float16 v16h;
typedef __attribute__((ext_vector_type(8)))  _Float16 v8h;
typedef __attribute__((ext_vector_type(8)))  float    v8f;
typedef __attribute__((ext_vector_type(4)))  unsigned int v4u;
typedef __attribute__((ext_vector_type(8)))  int      v8i;
typedef __attribute__((ext_vector_type(4)))  int      v4i;

#define D_MODEL 512
#define SEQ     1024
#define BATCH   16
#define M_TOT   (BATCH*SEQ)   // 16384 rows
#define GHEADS  128           // B*H pseudo-heads after raw reshape
#define SCALE_Q 0.35355339059327373f   // (dph//H)**-0.5 = 8**-0.5

// ---------------------------------------------------------------------------
// WMMA helpers.  16x16x32 f16 -> f32.
//   A (16x32 f16): lane = m + 16*b ; half h<8 -> K=8b+h ; h>=8 -> K=16+8b+(h-8)
//   B (32x16 f16): lane = n + 16*c ; half h<8 -> K=8c+h ; h>=8 -> K=16+8c+(h-8) (lane holds column n)
//   C (16x16 f32): lane = n + 16*(m>=8) ; vgpr r -> m = (lane>>4)*8 + r
// ---------------------------------------------------------------------------
__device__ __forceinline__ v8f wmma_f16(v16h a, v16h b, v8f c) {
    return __builtin_amdgcn_wmma_f32_16x16x32_f16(
        false, a, false, b, (short)0, c, false, false);
}

__device__ __forceinline__ v16h ldA(const _Float16* base, int stride, int colBase) {
    const int lane = threadIdx.x & 31;
    const int m = lane & 15, b = lane >> 4;
    const _Float16* p = base + m * stride + colBase;
    v8h lo = *(const v8h*)(p + 8 * b);
    v8h hi = *(const v8h*)(p + 16 + 8 * b);
    v16h r;
#pragma unroll
    for (int i = 0; i < 8; ++i) { r[i] = lo[i]; r[i + 8] = hi[i]; }
    return r;
}

// source is laid out [n][k] row-major with given stride (B pre-transposed)
__device__ __forceinline__ v16h ldB(const _Float16* base, int stride, int colBase) {
    const int lane = threadIdx.x & 31;
    const int n = lane & 15, c = lane >> 4;
    const _Float16* p = base + n * stride + colBase;
    v8h lo = *(const v8h*)(p + 8 * c);
    v8h hi = *(const v8h*)(p + 16 + 8 * c);
    v16h r;
#pragma unroll
    for (int i = 0; i < 8; ++i) { r[i] = lo[i]; r[i + 8] = hi[i]; }
    return r;
}

// ---------------------------------------------------------------------------
// Staging helpers (128-thread blocks, tile A 32x32 @ LDS stride 40,
// tile W 32(k) x 64(n) transposed -> Wt[n][k] stride 40).
// ---------------------------------------------------------------------------
__device__ __forceinline__ void stage_w(const float* __restrict__ W, _Float16* Wt,
                                        int kb, int n0) {
    const int tid = threadIdx.x;
    const int c  = tid & 63;         // output column n
    const int r0 = (tid >> 6) * 16;  // k sub-range
    v8h w0, w1;
#pragma unroll
    for (int r = 0; r < 8; ++r) {    // 16 independent coalesced loads -> clause
        w0[r] = (_Float16)W[(size_t)(kb + r0 + r) * D_MODEL + n0 + c];
        w1[r] = (_Float16)W[(size_t)(kb + r0 + 8 + r) * D_MODEL + n0 + c];
    }
    *(v8h*)(Wt + c * 40 + r0)     = w0;
    *(v8h*)(Wt + c * 40 + r0 + 8) = w1;
}

__device__ __forceinline__ void stage_a_f32(const float* __restrict__ X, _Float16* As,
                                            int m0, int kb) {
    const int tid = threadIdx.x;
    const int r = tid >> 2, c8 = (tid & 3) * 8;
    const float4* px = (const float4*)(X + (size_t)(m0 + r) * D_MODEL + kb + c8);
    float4 x0 = px[0], x1 = px[1];
    v8h a;
    a[0] = (_Float16)x0.x; a[1] = (_Float16)x0.y;
    a[2] = (_Float16)x0.z; a[3] = (_Float16)x0.w;
    a[4] = (_Float16)x1.x; a[5] = (_Float16)x1.y;
    a[6] = (_Float16)x1.z; a[7] = (_Float16)x1.w;
    *(v8h*)(As + r * 40 + c8) = a;
}

// ---------------------------------------------------------------------------
// Kernel 1: Y_f16 = (X_f32 @ W_f32 + bias) * scaleF          [16384,512]
// ---------------------------------------------------------------------------
__global__ __launch_bounds__(128) void proj_kernel(
    const float* __restrict__ X, const float* __restrict__ W,
    const float* __restrict__ bias, _Float16* __restrict__ Y, float scaleF)
{
    __shared__ _Float16 As[32 * 40];
    __shared__ _Float16 Wt[64 * 40];
    const int m0  = blockIdx.x * 32;
    const int n0  = blockIdx.y * 64;
    const int tid = threadIdx.x;
    const int wid = tid >> 5;
    const int mw  = (wid & 1) * 16;
    const int nw  = (wid >> 1) * 32;
    v8f acc0 = {}; v8f acc1 = {};

    for (int kb = 0; kb < D_MODEL; kb += 32) {
        __syncthreads();
        stage_a_f32(X, As, m0, kb);
        stage_w(W, Wt, kb, n0);
        __syncthreads();
        v16h a  = ldA(As + mw * 40, 40, 0);
        v16h b0 = ldB(Wt + nw * 40, 40, 0);
        v16h b1 = ldB(Wt + (nw + 16) * 40, 40, 0);
        acc0 = wmma_f16(a, b0, acc0);
        acc1 = wmma_f16(a, b1, acc1);
    }
    const int lane = tid & 31;
    const int nl = lane & 15, mh = lane >> 4;
#pragma unroll
    for (int r = 0; r < 8; ++r) {
        int row = m0 + mw + mh * 8 + r;
        int c0  = n0 + nw + nl;
        Y[(size_t)row * D_MODEL + c0]      = (_Float16)((acc0[r] + bias[c0]) * scaleF);
        Y[(size_t)row * D_MODEL + c0 + 16] = (_Float16)((acc1[r] + bias[c0 + 16]) * scaleF);
    }
}

// ---------------------------------------------------------------------------
// Kernel 2: attention for one (g, 128-row query block).
// K_g staged into LDS via the Tensor Data Mover (overlapped with manual
// transposed staging of V_g); two-pass softmax with running block max.
// ---------------------------------------------------------------------------
__global__ __launch_bounds__(256) void attn_kernel(
    const _Float16* __restrict__ qh, const _Float16* __restrict__ kh,
    const _Float16* __restrict__ vh, float* __restrict__ attnOut,
    _Float16* __restrict__ ch)
{
    extern __shared__ __align__(16) _Float16 smem[];
    _Float16* Kls = smem;                 // [1024][64] row-major
    _Float16* Vt  = smem + 65536;         // [64][1024] transposed
    const int g    = blockIdx.y;
    const int tid  = threadIdx.x;
    const int wid  = tid >> 5;
    const int lane = tid & 31;
    _Float16* Pst = smem + 131072 + wid * 512;   // wave-private [16][32]

    const _Float16* kg = kh + (size_t)g * 65536;
    const _Float16* vg = vh + (size_t)g * 65536;

    if (wid == 0) {   // one TDM op per workgroup: 1024x64 f16 tile, contiguous
        unsigned long long ga = (unsigned long long)(uintptr_t)kg;
        unsigned lds_k = (unsigned)(uintptr_t)(void*)Kls;  // low 32 bits = LDS byte addr
        v4u g0;
        g0[0] = 1u;                                         // count=1 (valid user D#)
        g0[1] = lds_k;                                      // lds_addr
        g0[2] = (unsigned)(ga & 0xffffffffu);               // global_addr[31:0]
        g0[3] = (unsigned)((ga >> 32) & 0x01ffffffu) | (2u << 30);  // addr[56:32] | type=2
        v8i g1;
        g1[0] = (int)(1u << 16);     // workgroup_mask=0, data_size=1 (2 bytes)
        g1[1] = (int)(64u << 16);    // tensor_dim0 = 64   (bits 79:48, low half)
        g1[2] = (int)(1024u << 16);  // tensor_dim1 = 1024 (bits 111:80, low half)
        g1[3] = (int)(64u << 16);    // tile_dim0 = 64     (bits 127:112)
        g1[4] = 1024;                // tile_dim1 = 1024, tile_dim2 = 0
        g1[5] = 64;                  // tensor_dim0_stride = 64 (48-bit, low dword)
        g1[6] = 0;
        g1[7] = 0;                   // tensor_dim1_stride = 0 (2-D tile)
        v4i gz4 = {0, 0, 0, 0};
        v8i gz8 = {0, 0, 0, 0, 0, 0, 0, 0};
        __builtin_amdgcn_tensor_load_to_lds(g0, g1, gz4, gz4, gz8, 0);
    }

    for (int i = tid; i < 8192; i += 256) {               // V transposed staging
        int s = i >> 3, db = (i & 7) * 8;
        v8h v = *(const v8h*)(vg + s * 64 + db);
#pragma unroll
        for (int j = 0; j < 8; ++j) Vt[(db + j) * 1024 + s] = v[j];
    }

    if (wid == 0) __builtin_amdgcn_s_wait_tensorcnt(0);
    __syncthreads();

    const int m0 = blockIdx.x * 128 + wid * 16;           // query row within g
    const _Float16* qg = qh + (size_t)g * 65536 + (size_t)m0 * 64;
    v16h aq0 = ldA(qg, 64, 0);
    v16h aq1 = ldA(qg, 64, 32);

    const int mh = lane >> 4;
    const int nl = lane & 15;
    float M = -1e30f;          // running block max (all 16 rows of this wave)
    float rsum[8];
#pragma unroll
    for (int r = 0; r < 8; ++r) rsum[r] = 0.0f;

    // ---- pass 1: running block max + exact per-row sums ----
    for (int jt = 0; jt < 64; ++jt) {
        const _Float16* kb = Kls + jt * 16 * 64;
        v8f c = {};
        c = wmma_f16(aq0, ldB(kb, 64, 0),  c);
        c = wmma_f16(aq1, ldB(kb, 64, 32), c);
        float tmax = c[0];
#pragma unroll
        for (int r = 1; r < 8; ++r) tmax = fmaxf(tmax, c[r]);
        tmax = fmaxf(tmax, __shfl_xor(tmax, 1, 32));
        tmax = fmaxf(tmax, __shfl_xor(tmax, 2, 32));
        tmax = fmaxf(tmax, __shfl_xor(tmax, 4, 32));
        tmax = fmaxf(tmax, __shfl_xor(tmax, 8, 32));
        tmax = fmaxf(tmax, __shfl_xor(tmax, 16, 32));
        float mnew = fmaxf(M, tmax);
        float corr = __expf(M - mnew);
#pragma unroll
        for (int r = 0; r < 8; ++r) {
            float e = __expf(c[r] - mnew);
            e += __shfl_xor(e, 1, 16);
            e += __shfl_xor(e, 2, 16);
            e += __shfl_xor(e, 4, 16);
            e += __shfl_xor(e, 8, 16);
            rsum[r] = rsum[r] * corr + e;
        }
        M = mnew;
    }
    float rinv[8];
#pragma unroll
    for (int r = 0; r < 8; ++r) rinv[r] = 1.0f / rsum[r];

    // ---- pass 2: write normalized P, accumulate context = P @ V ----
    v8f acc[4];
#pragma unroll
    for (int t = 0; t < 4; ++t) acc[t] = (v8f){};
    float* arow = attnOut + (size_t)g * SEQ * SEQ + (size_t)m0 * SEQ;

    for (int jp = 0; jp < 32; ++jp) {                     // pairs of 16-col tiles
#pragma unroll
        for (int t = 0; t < 2; ++t) {
            int jt = jp * 2 + t;
            const _Float16* kb = Kls + jt * 16 * 64;
            v8f c = {};
            c = wmma_f16(aq0, ldB(kb, 64, 0),  c);
            c = wmma_f16(aq1, ldB(kb, 64, 32), c);
#pragma unroll
            for (int r = 0; r < 8; ++r) {
                float p = __expf(c[r] - M) * rinv[r];
                arow[(size_t)(mh * 8 + r) * SEQ + jt * 16 + nl] = p;
                Pst[(mh * 8 + r) * 32 + t * 16 + nl] = (_Float16)p;
            }
        }
        v16h ap = ldA(Pst, 32, 0);                        // P re-layout C->A via LDS
#pragma unroll
        for (int t = 0; t < 4; ++t)
            acc[t] = wmma_f16(ap, ldB(Vt + t * 16 * 1024, 1024, jp * 32), acc[t]);
    }

    _Float16* cg = ch + (size_t)g * 65536 + (size_t)m0 * 64;
#pragma unroll
    for (int t = 0; t < 4; ++t)
#pragma unroll
        for (int r = 0; r < 8; ++r)
            cg[(mh * 8 + r) * 64 + t * 16 + nl] = (_Float16)acc[t][r];
}

// ---------------------------------------------------------------------------
// Kernel 3a: Y_f32 = context_f16 @ Wo + bo + residual      [16384,512]
// ---------------------------------------------------------------------------
__global__ __launch_bounds__(128) void outproj_kernel(
    const _Float16* __restrict__ Xh, const float* __restrict__ W,
    const float* __restrict__ bias, const float* __restrict__ resid,
    float* __restrict__ Y)
{
    __shared__ _Float16 As[32 * 40];
    __shared__ _Float16 Wt[64 * 40];
    const int m0  = blockIdx.x * 32;
    const int n0  = blockIdx.y * 64;
    const int tid = threadIdx.x;
    const int wid = tid >> 5;
    const int mw  = (wid & 1) * 16;
    const int nw  = (wid >> 1) * 32;
    v8f acc0 = {}; v8f acc1 = {};

    for (int kb = 0; kb < D_MODEL; kb += 32) {
        __syncthreads();
        {   // A tile copy (already f16): one b128 load + b128 LDS store / thread
            int r = tid >> 2, c8 = (tid & 3) * 8;
            *(v8h*)(As + r * 40 + c8) =
                *(const v8h*)(Xh + (size_t)(m0 + r) * D_MODEL + kb + c8);
        }
        stage_w(W, Wt, kb, n0);
        __syncthreads();
        v16h a  = ldA(As + mw * 40, 40, 0);
        v16h b0 = ldB(Wt + nw * 40, 40, 0);
        v16h b1 = ldB(Wt + (nw + 16) * 40, 40, 0);
        acc0 = wmma_f16(a, b0, acc0);
        acc1 = wmma_f16(a, b1, acc1);
    }
    const int lane = tid & 31;
    const int nl = lane & 15, mh = lane >> 4;
#pragma unroll
    for (int r = 0; r < 8; ++r) {
        int row = m0 + mw + mh * 8 + r;
        int c0  = n0 + nw + nl;
        Y[(size_t)row * D_MODEL + c0] =
            acc0[r] + bias[c0] + resid[(size_t)row * D_MODEL + c0];
        Y[(size_t)row * D_MODEL + c0 + 16] =
            acc1[r] + bias[c0 + 16] + resid[(size_t)row * D_MODEL + c0 + 16];
    }
}

// ---------------------------------------------------------------------------
// Kernel 3b: in-place LayerNorm per row of 512.
// ---------------------------------------------------------------------------
__global__ __launch_bounds__(256) void ln_kernel(
    float* __restrict__ Y, const float* __restrict__ gamma,
    const float* __restrict__ beta)
{
    const int row = blockIdx.x;
    float* y = Y + (size_t)row * D_MODEL;
    const int tid = threadIdx.x;
    float x0 = y[tid], x1 = y[tid + 256];
    float s = x0 + x1;
    float q = x0 * x0 + x1 * x1;
#pragma unroll
    for (int off = 16; off >= 1; off >>= 1) {
        s += __shfl_xor(s, off, 32);
        q += __shfl_xor(q, off, 32);
    }
    __shared__ float ws[8], wq[8];
    const int wid = tid >> 5, lane = tid & 31;
    if (lane == 0) { ws[wid] = s; wq[wid] = q; }
    __syncthreads();
    if (tid == 0) {
        float S = 0.f, Q = 0.f;
#pragma unroll
        for (int i = 0; i < 8; ++i) { S += ws[i]; Q += wq[i]; }
        ws[0] = S; wq[0] = Q;
    }
    __syncthreads();
    float mu  = ws[0] * (1.0f / 512.0f);
    float var = wq[0] * (1.0f / 512.0f) - mu * mu;
    float rs  = rsqrtf(var + 1e-5f);
    y[tid]       = (x0 - mu) * rs * gamma[tid] + beta[tid];
    y[tid + 256] = (x1 - mu) * rs * gamma[tid + 256] + beta[tid + 256];
}

// ---------------------------------------------------------------------------
extern "C" void kernel_launch(void* const* d_in, const int* in_sizes, int n_in,
                              void* d_out, int out_size, void* d_ws, size_t ws_size,
                              hipStream_t stream) {
    (void)in_sizes; (void)n_in; (void)out_size; (void)ws_size;
    const float* key_in   = (const float*)d_in[0];
    const float* value_in = (const float*)d_in[1];
    const float* query_in = (const float*)d_in[2];
    const float* Wk = (const float*)d_in[3];
    const float* bk = (const float*)d_in[4];
    const float* Wq = (const float*)d_in[5];
    const float* bq = (const float*)d_in[6];
    const float* Wv = (const float*)d_in[7];
    const float* bv = (const float*)d_in[8];
    const float* Wo = (const float*)d_in[9];
    const float* bo = (const float*)d_in[10];
    const float* gamma = (const float*)d_in[11];
    const float* beta  = (const float*)d_in[12];

    float* outY  = (float*)d_out;                          // [16384,512]
    float* attnY = outY + (size_t)M_TOT * D_MODEL;         // [128,1024,1024]

    const size_t N = (size_t)M_TOT * D_MODEL;
    _Float16* qh  = (_Float16*)d_ws;
    _Float16* khp = qh + N;
    _Float16* vhp = khp + N;
    _Float16* chp = vhp + N;                               // total 64 MB f16

    dim3 gproj(M_TOT / 32, D_MODEL / 64);                  // 512 x 8
    proj_kernel<<<gproj, 128, 0, stream>>>(query_in, Wq, bq, qh, SCALE_Q);
    proj_kernel<<<gproj, 128, 0, stream>>>(key_in,   Wk, bk, khp, 1.0f);
    proj_kernel<<<gproj, 128, 0, stream>>>(value_in, Wv, bv, vhp, 1.0f);

    size_t shmem = (size_t)(65536 + 65536 + 8 * 512) * sizeof(_Float16); // 270336B
    attn_kernel<<<dim3(SEQ / 128, GHEADS), 256, shmem, stream>>>(qh, khp, vhp, attnY, chp);

    outproj_kernel<<<gproj, 128, 0, stream>>>(chp, Wo, bo, query_in, outY);
    ln_kernel<<<M_TOT, 256, 0, stream>>>(outY, gamma, beta);
}
